// LED_72679436583504
// MI455X (gfx1250) — matrix operand order repaired and checked by reading, not compiled, not run on hardware
//
#include <hip/hip_runtime.h>
#include <hip/hip_bf16.h>
#include <float.h>

typedef __attribute__((ext_vector_type(16))) _Float16 v16h;
typedef __attribute__((ext_vector_type(8)))  float    v8f;

#define HW    512
#define NPIX  (HW * HW)          // 262144
#define NB    8
#define NCH   11
#define BINS  32
#define EPS_F 1e-6f

// combinations(range(5), 2) in reference order; index 8 == (2,4)
__constant__ int PAIR_A[10] = {0,0,0,0,1,1,1,2,2,3};
__constant__ int PAIR_B[10] = {1,2,3,4,2,3,4,3,4,4};

// ---------------------------------------------------------------------------
// Pass 1: per-(b, pair) min/max of channel diffs. One block per (b, pair).
// ---------------------------------------------------------------------------
__global__ __launch_bounds__(256)
void diff_minmax_kernel(const float* __restrict__ x,
                        float* __restrict__ mins, float* __restrict__ maxs) {
    int blk = blockIdx.x;
    int b = blk / 10, c = blk % 10;
    const float* xa = x + ((size_t)b * 5 + PAIR_A[c]) * NPIX;
    const float* xb = x + ((size_t)b * 5 + PAIR_B[c]) * NPIX;
    float mn = FLT_MAX, mx = -FLT_MAX;
    for (int p = threadIdx.x; p < NPIX; p += 256) {
        float v = xa[p] - xb[p];
        mn = fminf(mn, v);
        mx = fmaxf(mx, v);
    }
    __shared__ float smn[256], smx[256];
    smn[threadIdx.x] = mn; smx[threadIdx.x] = mx;
    __syncthreads();
    for (int s = 128; s > 0; s >>= 1) {
        if (threadIdx.x < s) {
            smn[threadIdx.x] = fminf(smn[threadIdx.x], smn[threadIdx.x + s]);
            smx[threadIdx.x] = fmaxf(smx[threadIdx.x], smx[threadIdx.x + s]);
        }
        __syncthreads();
    }
    if (threadIdx.x == 0) {
        mins[b * NCH + c] = smn[0];
        maxs[b * NCH + c] = smx[0];
    }
}

// ---------------------------------------------------------------------------
// Pass 2: bin index (u8) for diff-channel 8 = x[:,2] - x[:,4].
// ---------------------------------------------------------------------------
__global__ __launch_bounds__(256)
void binidx_kernel(const float* __restrict__ x,
                   const float* __restrict__ mins, const float* __restrict__ maxs,
                   unsigned char* __restrict__ bidx) {
    int i = blockIdx.x * 256 + threadIdx.x;
    if (i >= NB * NPIX) return;
    int b = i / NPIX, p = i % NPIX;
    float v  = x[((size_t)b * 5 + 2) * NPIX + p] - x[((size_t)b * 5 + 4) * NPIX + p];
    float mn = mins[b * NCH + 8], mx = maxs[b * NCH + 8];
    float r  = (v - mn) / (mx - mn + EPS_F);
    int idx  = (int)(r * (float)(BINS - 1));   // truncation, matches .astype(int32)
    idx = idx < 0 ? 0 : (idx > 31 ? 31 : idx);
    bidx[i] = (unsigned char)idx;
}

// ---------------------------------------------------------------------------
// Pass 3: local entropy via WMMA. One wave per 16x16 tile.
//
// Vertical 9-tap: sliding per-column bin counts (registers, fed from a 24x24
// halo of bin indices in LDS, column-major).
// Horizontal 9-tap: hist[bin][x] = sum_k V[bin][k] * Band[k][x], done as two
// v_wmma_f32_16x16x32_f16 (bins 0-15 and 16-31 as M, 24 halo cols as K
// padded to 32, 16 output cols as N). Counts <= 81 are exact in f16.
//
// A fragment layout (ISA 7.12.2, 16-bit A 16x32): M = lane&15,
//   half h -> K = h + (h>=8 ? 8 : 0) + (lane>=16 ? 8 : 0)
// B fragment layout (16-bit B 32x16): N = lane&15,
//   half h -> K = h + (h>=8 ? 8 : 0) + (lane>=16 ? 8 : 0)
// D layout: lane<16 -> (M=r, N=lane); lane>=16 -> (M=r+8, N=lane-16).
// So column n's 32 bins live split across lanes n and n+16 -> combine with
// __shfl_xor(.., 16) on wave32.
// ---------------------------------------------------------------------------
__global__ __launch_bounds__(32)
void entropy_kernel(const unsigned char* __restrict__ bidx,
                    float* __restrict__ ent) {
    __shared__ unsigned int lidx[24 * 24];   // column-major: [col*24 + row]

    int tile = blockIdx.x;
    int b  = tile >> 10;                     // 32x32 = 1024 tiles per image
    int t  = tile & 1023;
    int ty0 = (t >> 5) * 16;
    int tx0 = (t & 31) * 16;
    const unsigned char* img = bidx + (size_t)b * NPIX;
    int lane = threadIdx.x;

    // load 24x24 halo; out-of-image -> sentinel 255 (matches zero padding of
    // the one-hot in the reference: contributes to no bin, shrinks tot)
    for (int i = lane; i < 24 * 24; i += 32) {
        int col = i / 24, row = i % 24;
        int gx = tx0 + col - 4, gy = ty0 + row - 4;
        unsigned int v = 255u;
        if (gx >= 0 && gx < HW && gy >= 0 && gy < HW)
            v = img[gy * HW + gx];
        lidx[i] = v;
    }
    __syncthreads();

    int hi = lane >> 4;
    int n  = lane & 15;
    unsigned int bin0 = (unsigned int)n;        // A0 covers bins 0..15
    unsigned int bin1 = (unsigned int)(n + 16); // A1 covers bins 16..31

    // constant band matrix fragment: B[k][n] = 1 iff 0 <= k-n <= 8
    v16h bf;
#pragma unroll
    for (int h = 0; h < 16; ++h) {
        int k = h + ((h >> 3) << 3) + hi * 8;
        int d = k - n;
        bf[h] = (_Float16)((d >= 0 && d <= 8) ? 1.0f : 0.0f);
    }

    // slot -> K mapping for the A fragment; K in [0,24) are real halo columns
    int kk[16];
    bool valid[16];
    int cnt0[16], cnt1[16];
#pragma unroll
    for (int s = 0; s < 16; ++s) {
        int k = s + ((s >> 3) << 3) + hi * 8;
        kk[s] = k;
        valid[s] = (k < 24);
        int c0 = 0, c1 = 0;
        if (valid[s]) {
#pragma unroll
            for (int h = 0; h < 9; ++h) {        // init window for y = 0
                unsigned int u = lidx[k * 24 + h];
                c0 += (u == bin0);
                c1 += (u == bin1);
            }
        }
        cnt0[s] = c0; cnt1[s] = c1;
    }

    float* eimg = ent + (size_t)b * NPIX;
    for (int y = 0; y < 16; ++y) {
        v16h a0, a1;
#pragma unroll
        for (int s = 0; s < 16; ++s) {
            a0[s] = (_Float16)cnt0[s];
            a1[s] = (_Float16)cnt1[s];
        }
        v8f z = {};
        v8f d0 = __builtin_amdgcn_wmma_f32_16x16x32_f16(
            false, a0, false, bf, (short)0, z, false, false);
        v8f d1 = __builtin_amdgcn_wmma_f32_16x16x32_f16(
            false, a1, false, bf, (short)0, z, false, false);

        // entropy: ent = log2(T) - (1/T) * sum_c c*log2(c); zero-count terms
        // contribute 0, matching -p*log2(max(p,eps)) with p = 0.
        float S = 0.0f, T = 0.0f;
#pragma unroll
        for (int r = 0; r < 8; ++r) {
            float c0 = d0[r], c1 = d1[r];
            T += c0 + c1;
            S += (c0 > 0.5f) ? c0 * __log2f(c0) : 0.0f;
            S += (c1 > 0.5f) ? c1 * __log2f(c1) : 0.0f;
        }
        S += __shfl_xor(S, 16, 32);   // combine the two lanes holding column n
        T += __shfl_xor(T, 16, 32);   // T >= 25 always (5x5 corner window)
        float e = __log2f(T) - S / T;
        if (lane < 16)
            eimg[(size_t)(ty0 + y) * HW + tx0 + n] = e;

        if (y < 15) {                 // slide window: drop halo row y, add y+9
#pragma unroll
            for (int s = 0; s < 16; ++s) {
                if (valid[s]) {
                    unsigned int uo = lidx[kk[s] * 24 + y];
                    unsigned int un = lidx[kk[s] * 24 + y + 9];
                    cnt0[s] += (int)(un == bin0) - (int)(uo == bin0);
                    cnt1[s] += (int)(un == bin1) - (int)(uo == bin1);
                }
            }
        }
    }
}

// ---------------------------------------------------------------------------
// Pass 4: per-image min/max of the entropy map (channel 10). One block/image.
// ---------------------------------------------------------------------------
__global__ __launch_bounds__(256)
void ent_minmax_kernel(const float* __restrict__ ent,
                       float* __restrict__ mins, float* __restrict__ maxs) {
    int b = blockIdx.x;
    const float* e = ent + (size_t)b * NPIX;
    float mn = FLT_MAX, mx = -FLT_MAX;
    for (int p = threadIdx.x; p < NPIX; p += 256) {
        float v = e[p];
        mn = fminf(mn, v);
        mx = fmaxf(mx, v);
    }
    __shared__ float smn[256], smx[256];
    smn[threadIdx.x] = mn; smx[threadIdx.x] = mx;
    __syncthreads();
    for (int s = 128; s > 0; s >>= 1) {
        if (threadIdx.x < s) {
            smn[threadIdx.x] = fminf(smn[threadIdx.x], smn[threadIdx.x + s]);
            smx[threadIdx.x] = fmaxf(smx[threadIdx.x], smx[threadIdx.x + s]);
        }
        __syncthreads();
    }
    if (threadIdx.x == 0) {
        mins[b * NCH + 10] = smn[0];
        maxs[b * NCH + 10] = smx[0];
    }
}

// ---------------------------------------------------------------------------
// Pass 5: recompute diffs (L2-resident reads), normalize, write all channels.
// ---------------------------------------------------------------------------
__global__ __launch_bounds__(256)
void finalize_kernel(const float* __restrict__ x, const float* __restrict__ ent,
                     const float* __restrict__ mins, const float* __restrict__ maxs,
                     float* __restrict__ out, int total) {
    int i = blockIdx.x * 256 + threadIdx.x;
    if (i >= total) return;
    int p  = i % NPIX;
    int bc = i / NPIX;
    int c  = bc % NCH;
    int b  = bc / NCH;
    float v;
    if (c < 10) {
        v = x[((size_t)b * 5 + PAIR_A[c]) * NPIX + p]
          - x[((size_t)b * 5 + PAIR_B[c]) * NPIX + p];
    } else {
        v = ent[(size_t)b * NPIX + p];
    }
    float mn = mins[bc], mx = maxs[bc];
    out[i] = (v - mn) / (mx - mn + EPS_F);
}

// ---------------------------------------------------------------------------
// Host launch. Workspace layout (needs ~10.5 MB):
//   [0       .. 352)    mins[88]
//   [512     .. 864)    maxs[88]
//   [4096    .. +2MB)   u8 bin indices, 8*512*512
//   [4096+2MB.. +8MB)   f32 entropy maps, 8*512*512
// ---------------------------------------------------------------------------
extern "C" void kernel_launch(void* const* d_in, const int* in_sizes, int n_in,
                              void* d_out, int out_size, void* d_ws, size_t ws_size,
                              hipStream_t stream) {
    (void)in_sizes; (void)n_in; (void)out_size; (void)ws_size;
    const float* x = (const float*)d_in[0];
    float* out = (float*)d_out;
    char* ws = (char*)d_ws;
    float* mins = (float*)(ws);
    float* maxs = (float*)(ws + 512);
    unsigned char* bidx = (unsigned char*)(ws + 4096);
    float* ent = (float*)(ws + 4096 + (size_t)NB * NPIX);

    diff_minmax_kernel<<<NB * 10, 256, 0, stream>>>(x, mins, maxs);

    int tot_px = NB * NPIX;
    binidx_kernel<<<(tot_px + 255) / 256, 256, 0, stream>>>(x, mins, maxs, bidx);

    entropy_kernel<<<NB * 32 * 32, 32, 0, stream>>>(bidx, ent);

    ent_minmax_kernel<<<NB, 256, 0, stream>>>(ent, mins, maxs);

    int total = NB * NCH * NPIX;
    finalize_kernel<<<(total + 255) / 256, 256, 0, stream>>>(x, ent, mins, maxs, out, total);
}
